// Attention_v2_cross_21603685499612
// MI455X (gfx1250) — compile-verified, matrix operand order checked by reading, not run on hardware
//
#include <hip/hip_runtime.h>
#include <hip/hip_bf16.h>
#include <math.h>
#include <stdint.h>

typedef __bf16 bf16_t;
typedef __attribute__((ext_vector_type(16))) __bf16 v16bf;
typedef __attribute__((ext_vector_type(8)))  __bf16 v8bf;
typedef __attribute__((ext_vector_type(8)))  float  v8f;
typedef __attribute__((ext_vector_type(4)))  unsigned int u32x4;
typedef __attribute__((ext_vector_type(8)))  int i32x8;
typedef __attribute__((ext_vector_type(4)))  int i32x4;

#define BB     4
#define NN     1024
#define RR     12
#define DIM    512
#define HEADS  8
#define DHEAD  64
#define RD     (RR*DHEAD)      // 768 (contraction for sim / out)
#define MROWS  (BB*NN*RR)      // 49152 token rows
#define JT     1056            // padded (n+1)=1025 -> 66 tiles of 16
#define NVALID 1025
#define QSCALE 2.819092e-4f    // 64^-0.5 / 128 / sqrt(12)

#if defined(__has_builtin)
#if __has_builtin(__builtin_amdgcn_tensor_load_to_lds) && __has_builtin(__builtin_amdgcn_s_wait_tensorcnt)
#define HAVE_TDM 1
#endif
#endif
#ifndef HAVE_TDM
#define HAVE_TDM 0
#endif

// ---------------- WMMA helpers (CDNA5 v_wmma_f32_16x16x32_bf16) -------------
// 16-bit operand layout (ISA 7.12.2): lane%16 = row, lane/16 selects k-half;
// elements e -> k = (e%8) + 16*(e/8) + 8*(lane/16)  => two contiguous 16B chunks.
__device__ __forceinline__ v16bf load_frag(const bf16_t* base, int stride, int lane) {
    int r  = lane & 15;
    int kh = lane >> 4;
    const bf16_t* p = base + (size_t)r * stride + 8 * kh;
    union { v16bf v; v8bf h[2]; } u;
    u.h[0] = *reinterpret_cast<const v8bf*>(p);
    u.h[1] = *reinterpret_cast<const v8bf*>(p + 16);
    return u.v;
}

__device__ __forceinline__ v8f wmma_bf16(v16bf a, v16bf b, v8f c) {
    return __builtin_amdgcn_wmma_f32_16x16x32_bf16(false, a, false, b, (short)0, c, false, false);
}

// ---------------- TDM 1-D tile stage: global -> LDS (ISA ch.8 D#) -----------
#if HAVE_TDM
__device__ __forceinline__ void tdm_load_1d(void* ldsDst, const void* gSrc, unsigned bytes) {
    unsigned n = bytes >> 3;                 // elements of 8 bytes, fits tile_dim0 (16b)
    unsigned long long ga = (unsigned long long)(uintptr_t)gSrc;
    u32x4 g0 = {};
    g0[0] = 1u;                              // count=1, is_restore=0, gather off
    g0[1] = (unsigned)(uintptr_t)ldsDst;     // lds_addr = low 32 bits of flat LDS ptr
    g0[2] = (unsigned)ga;                    // global_addr[31:0]
    g0[3] = (unsigned)(ga >> 32) | (2u << 30); // global_addr[56:32] | type=2 ("image")
    i32x8 g1 = {};
    g1[0] = (int)(3u << 16);                 // workgroup_mask=0, data_size=3 (8B)
    g1[1] = (int)(n << 16);                  // tensor_dim0[15:0] at bits 63:48
    g1[2] = (int)(1u << 16);                 // tensor_dim1 = 1
    g1[3] = (int)(n << 16);                  // tile_dim0 at bits 127:112 (tile_dim1=0 -> 1D)
    g1[5] = (int)n;                          // tensor_dim0_stride low 32
    i32x4 z4 = {};
#if __clang_major__ >= 23
    i32x8 z8 = {};
    __builtin_amdgcn_tensor_load_to_lds(g0, g1, z4, z4, z8, 0);
#else
    __builtin_amdgcn_tensor_load_to_lds(g0, g1, z4, z4, 0);
#endif
}
#endif

__device__ __forceinline__ void stage_tile(void* ldsDst, const void* gSrc, unsigned bytes, int t) {
#if HAVE_TDM
    if ((t >> 5) == 0) {                     // wave 0 issues the DMA
        tdm_load_1d(ldsDst, gSrc, bytes);
        __builtin_amdgcn_s_wait_tensorcnt(0);
    }
    __syncthreads();
#else
    const uint4* s = (const uint4*)gSrc;
    uint4* d = (uint4*)ldsDst;
    unsigned nv = bytes >> 4;
    for (unsigned i = t; i < nv; i += blockDim.x) d[i] = s[i];
    __syncthreads();
#endif
}

// ---------------- K0: cast weights to bf16 ----------------------------------
__global__ void k_cast_w(const float* __restrict__ wq, const float* __restrict__ wkv,
                         const float* __restrict__ wout,
                         bf16_t* __restrict__ bq, bf16_t* __restrict__ bkv,
                         bf16_t* __restrict__ bout) {
    int i = blockIdx.x * 256 + threadIdx.x;
    if (i < DIM * DIM)   bq[i]   = (bf16_t)wq[i];
    if (i < 2*DIM*DIM)   bkv[i]  = (bf16_t)wkv[i];
    if (i < DIM * DIM)   bout[i] = (bf16_t)wout[i];
}

// ---------------- K1: null-kv row + zero tail padding -----------------------
// k_t: [32][JT][RD] (j-major), v_t: [32][RD][JT] (rd-major, transposed)
__global__ void k_init_kv(const float* __restrict__ nkv,
                          bf16_t* __restrict__ k_t, bf16_t* __restrict__ v_t) {
    int idx = blockIdx.x * 256 + threadIdx.x;
    if (idx >= 32 * RD * 64) return;
    int slot = idx & 63;
    int rd   = (idx >> 6) % RD;
    int bh   = idx / (RD * 64);
    int d    = rd & 63;
    if (slot == 0)       k_t[((size_t)bh*JT + 0)*RD + rd]             = (bf16_t)nkv[d];
    else if (slot == 1)  v_t[((size_t)bh*RD + rd)*JT + 0]             = (bf16_t)nkv[64 + d];
    else if (slot < 33)  k_t[((size_t)bh*JT + (1023 + slot))*RD + rd] = (bf16_t)0.f;  // j=1025..1055
    else                 v_t[((size_t)bh*RD + rd)*JT + (992 + slot)]  = (bf16_t)0.f;  // j=1025..1055
}

// ---------------- K2: LayerNorm(x)->bf16 and context->bf16 ------------------
__global__ void k_ln_cast(const float* __restrict__ x, const float* __restrict__ ctx,
                          const float* __restrict__ g,
                          bf16_t* __restrict__ xn, bf16_t* __restrict__ cn) {
    int row = blockIdx.x;       // 0..MROWS-1
    int t   = threadIdx.x;      // 128 threads
    if (blockIdx.y == 1) {
        const float* src = ctx + (size_t)row * DIM;
        bf16_t* dst = cn + (size_t)row * DIM;
        for (int i = t; i < DIM; i += 128) dst[i] = (bf16_t)src[i];
        return;
    }
    __shared__ float red[128];
    const float* src = x + (size_t)row * DIM;
    float v0[4]; float s = 0.f;
    for (int i = 0; i < 4; ++i) { v0[i] = src[t + 128 * i]; s += v0[i]; }
    red[t] = s; __syncthreads();
    for (int off = 64; off > 0; off >>= 1) { if (t < off) red[t] += red[t + off]; __syncthreads(); }
    float mean = red[0] * (1.f / DIM);
    __syncthreads();
    float s2 = 0.f;
    for (int i = 0; i < 4; ++i) { float d = v0[i] - mean; s2 += d * d; }
    red[t] = s2; __syncthreads();
    for (int off = 64; off > 0; off >>= 1) { if (t < off) red[t] += red[t + off]; __syncthreads(); }
    float rstd = rsqrtf(red[0] * (1.f / DIM) + 1e-5f);
    bf16_t* dst = xn + (size_t)row * DIM;
    for (int i = 0; i < 4; ++i) { int c = t + 128 * i; dst[c] = (bf16_t)((v0[i] - mean) * rstd * g[c]); }
}

// ---------------- K3: Q projection (scale folded in) ------------------------
// block covers 16 rows x 128 cols; wave w: col tiles e0=y*128+w*16 and e0+64,
// sharing one A-fragment per k-step (2x register blocking on B).
__global__ void k_qproj(const bf16_t* __restrict__ xn, const bf16_t* __restrict__ wq,
                        bf16_t* __restrict__ q_t) {
    __shared__ bf16_t As[16 * DIM];
    int t = threadIdx.x, lane = t & 31, w = t >> 5;
    int m0 = blockIdx.x * 16;
    stage_tile(As, xn + (size_t)m0 * DIM, 16 * DIM * 2, t);

    int e0 = blockIdx.y * 128 + w * 16;
    int e1 = e0 + 64;
    v8f acc0 = {}, acc1 = {};
    for (int ks = 0; ks < DIM / 32; ++ks) {
        v16bf a  = load_frag(As + ks * 32, DIM, lane);
        v16bf b0 = load_frag(wq + (size_t)e0 * DIM + ks * 32, DIM, lane);
        v16bf b1 = load_frag(wq + (size_t)e1 * DIM + ks * 32, DIM, lane);
        acc0 = wmma_bf16(a, b0, acc0);
        acc1 = wmma_bf16(a, b1, acc1);
    }
    int jn = lane & 15, half = lane >> 4;
    int ea = e0 + jn, ha = ea >> 6, da = ea & 63;
    int eb = e1 + jn, hb = eb >> 6, db = eb & 63;
    for (int v = 0; v < 8; ++v) {
        int m = m0 + v + 8 * half;
        int b_ = m / (NN * RR); int mm = m - b_ * NN * RR; int n = mm / RR; int r = mm - n * RR;
        q_t[((size_t)(b_ * 8 + ha) * NN + n) * RD + r * 64 + da] = (bf16_t)(acc0[v] * QSCALE);
        q_t[((size_t)(b_ * 8 + hb) * NN + n) * RD + r * 64 + db] = (bf16_t)(acc1[v] * QSCALE);
    }
}

// ---------------- K4: KV projection -----------------------------------------
__global__ void k_kvproj(const bf16_t* __restrict__ cn, const bf16_t* __restrict__ wkv,
                         bf16_t* __restrict__ k_t, bf16_t* __restrict__ v_t) {
    __shared__ bf16_t As[16 * DIM];
    int t = threadIdx.x, lane = t & 31, w = t >> 5;
    int m0 = blockIdx.x * 16;
    stage_tile(As, cn + (size_t)m0 * DIM, 16 * DIM * 2, t);

    int e0 = blockIdx.y * 128 + w * 16;   // 0..1023
    int e1 = e0 + 64;
    v8f acc0 = {}, acc1 = {};
    for (int ks = 0; ks < DIM / 32; ++ks) {
        v16bf a  = load_frag(As + ks * 32, DIM, lane);
        v16bf b0 = load_frag(wkv + (size_t)e0 * DIM + ks * 32, DIM, lane);
        v16bf b1 = load_frag(wkv + (size_t)e1 * DIM + ks * 32, DIM, lane);
        acc0 = wmma_bf16(a, b0, acc0);
        acc1 = wmma_bf16(a, b1, acc1);
    }
    int jn = lane & 15, half = lane >> 4;
    for (int v = 0; v < 8; ++v) {
        int m = m0 + v + 8 * half;
        int b_ = m / (NN * RR); int mm = m - b_ * NN * RR; int n = mm / RR; int r = mm - n * RR;
        float vals[2] = { acc0[v], acc1[v] };
        int   es[2]   = { e0 + jn, e1 + jn };
        for (int q = 0; q < 2; ++q) {
            int e = es[q];
            if (e < DIM) {
                int h = e >> 6, d = e & 63;
                k_t[((size_t)(b_ * 8 + h) * JT + (n + 1)) * RD + r * 64 + d] = (bf16_t)vals[q];
            } else {
                int e2 = e - DIM; int h = e2 >> 6, d = e2 & 63;
                v_t[((size_t)(b_ * 8 + h) * RD + r * 64 + d) * JT + (n + 1)] = (bf16_t)vals[q];
            }
        }
    }
}

// ---------------- K5: fused attention (sim + softmax + P*V) -----------------
// block = 128 threads (4 waves), grid = (n/32, b*h). LDS: Q 48K | sim 132K | P 66K
// 2x2 register blocking: A-frags reused across 2 j/n tiles, B-frags across 2 i-tiles.
__global__ void k_attn(const bf16_t* __restrict__ q_t, const bf16_t* __restrict__ k_t,
                       const bf16_t* __restrict__ v_t, bf16_t* __restrict__ om) {
    extern __shared__ char smem[];
    bf16_t* Qs   = (bf16_t*)smem;                               // 32*768 bf16
    float*  simS = (float*)(smem + 32 * RD * 2);                // 32*1056 f32
    bf16_t* Ps   = (bf16_t*)(smem + 32 * RD * 2 + 32 * JT * 4); // 32*1056 bf16
    __shared__ float redm[128];
    __shared__ float reds[128];

    int t = threadIdx.x, lane = t & 31, w = t >> 5;
    int i0 = blockIdx.x * 32;
    int bh = blockIdx.y;

    stage_tile(Qs, q_t + ((size_t)bh * NN + i0) * RD, 32 * RD * 2, t);

    // ---- sim = Q * K^T : pairs of 16-wide j-tiles, 24 k-steps of 32 ----
    for (int jj = w * 2; jj < JT / 16; jj += 8) {
        v8f a00 = {}, a01 = {}, a10 = {}, a11 = {};
        const bf16_t* kb0 = k_t + ((size_t)bh * JT + jj * 16) * RD;
        const bf16_t* kb1 = kb0 + 16 * RD;
        for (int ks = 0; ks < RD / 32; ++ks) {
            v16bf a0 = load_frag(Qs + ks * 32, RD, lane);
            v16bf a1 = load_frag(Qs + 16 * RD + ks * 32, RD, lane);
            v16bf b0 = load_frag(kb0 + ks * 32, RD, lane);
            v16bf b1 = load_frag(kb1 + ks * 32, RD, lane);
            a00 = wmma_bf16(a0, b0, a00);
            a10 = wmma_bf16(a1, b0, a10);
            a01 = wmma_bf16(a0, b1, a01);
            a11 = wmma_bf16(a1, b1, a11);
        }
        int jn = lane & 15, half = lane >> 4;
        for (int v = 0; v < 8; ++v) {
            int i = v + 8 * half;
            simS[i * JT + jj * 16 + jn]              = a00[v];
            simS[i * JT + (jj + 1) * 16 + jn]        = a01[v];
            simS[(i + 16) * JT + jj * 16 + jn]       = a10[v];
            simS[(i + 16) * JT + (jj + 1) * 16 + jn] = a11[v];
        }
    }
    __syncthreads();

    // ---- pb-relax softmax: 4 threads/row, 264 cols each ----
    {
        int row = t >> 2, seg = t & 3;
        int c0 = seg * 264, c1 = c0 + 264;
        float mx = -INFINITY;
        for (int j = c0; j < c1; ++j) {
            float s = (j < NVALID) ? simS[row * JT + j] : -INFINITY;
            mx = fmaxf(mx, s);
        }
        redm[row * 4 + seg] = mx; __syncthreads();
        float rowmax = fmaxf(fmaxf(redm[row * 4], redm[row * 4 + 1]),
                             fmaxf(redm[row * 4 + 2], redm[row * 4 + 3]));
        float sum = 0.f;
        for (int j = c0; j < c1; ++j) {
            float s = (j < NVALID) ? simS[row * JT + j] : -INFINITY;
            float p = __expf((s - rowmax) * 128.0f);
            simS[row * JT + j] = p;
            sum += p;
        }
        reds[row * 4 + seg] = sum; __syncthreads();
        float tot = reds[row * 4] + reds[row * 4 + 1] + reds[row * 4 + 2] + reds[row * 4 + 3];
        float rinv = 1.0f / tot;
        for (int j = c0; j < c1; ++j)
            Ps[row * JT + j] = (bf16_t)(simS[row * JT + j] * rinv);
    }
    __syncthreads();

    // ---- out = P * V : pairs of 16-wide rd-tiles, 33 k-steps of 32 over j ----
    int b_ = bh >> 3, h = bh & 7;
    for (int nn = w * 2; nn < RD / 16; nn += 8) {
        v8f a00 = {}, a01 = {}, a10 = {}, a11 = {};
        const bf16_t* vb0 = v_t + ((size_t)bh * RD + nn * 16) * JT;
        const bf16_t* vb1 = vb0 + 16 * JT;
        for (int ks = 0; ks < JT / 32; ++ks) {
            v16bf a0 = load_frag(Ps + ks * 32, JT, lane);
            v16bf a1 = load_frag(Ps + 16 * JT + ks * 32, JT, lane);
            v16bf b0 = load_frag(vb0 + ks * 32, JT, lane);
            v16bf b1 = load_frag(vb1 + ks * 32, JT, lane);
            a00 = wmma_bf16(a0, b0, a00);
            a10 = wmma_bf16(a1, b0, a10);
            a01 = wmma_bf16(a0, b1, a01);
            a11 = wmma_bf16(a1, b1, a11);
        }
        int jn = lane & 15, half = lane >> 4;
        int rd0 = nn * 16 + jn,        r0 = rd0 >> 6, d0 = rd0 & 63;
        int rd1 = (nn + 1) * 16 + jn,  r1 = rd1 >> 6, d1 = rd1 & 63;
        for (int v = 0; v < 8; ++v) {
            int n = i0 + v + 8 * half;
            om[(((size_t)b_ * NN + n) * RR + r0) * DIM + h * 64 + d0] = (bf16_t)a00[v];
            om[(((size_t)b_ * NN + n) * RR + r1) * DIM + h * 64 + d1] = (bf16_t)a01[v];
            n += 16;
            om[(((size_t)b_ * NN + n) * RR + r0) * DIM + h * 64 + d0] = (bf16_t)a10[v];
            om[(((size_t)b_ * NN + n) * RR + r1) * DIM + h * 64 + d1] = (bf16_t)a11[v];
        }
    }
}

// ---------------- K6: out projection + output LayerNorm ---------------------
__global__ void k_outproj(const bf16_t* __restrict__ om, const bf16_t* __restrict__ wout,
                          const float* __restrict__ gout, float* __restrict__ out) {
    __shared__ bf16_t As[16 * DIM];
    __shared__ float  Cs[16 * DIM];
    __shared__ float  red[128];
    int t = threadIdx.x, lane = t & 31, w = t >> 5;
    int m0 = blockIdx.x * 16;
    stage_tile(As, om + (size_t)m0 * DIM, 16 * DIM * 2, t);

    for (int nn = w * 2; nn < DIM / 16; nn += 8) {
        v8f acc0 = {}, acc1 = {};
        for (int ks = 0; ks < DIM / 32; ++ks) {
            v16bf a  = load_frag(As + ks * 32, DIM, lane);
            v16bf b0 = load_frag(wout + (size_t)(nn * 16) * DIM + ks * 32, DIM, lane);
            v16bf b1 = load_frag(wout + (size_t)((nn + 1) * 16) * DIM + ks * 32, DIM, lane);
            acc0 = wmma_bf16(a, b0, acc0);
            acc1 = wmma_bf16(a, b1, acc1);
        }
        int o0 = nn * 16 + (lane & 15);
        int o1 = o0 + 16;
        for (int v = 0; v < 8; ++v) {
            int i = v + 8 * (lane >> 4);
            Cs[i * DIM + o0] = acc0[v];
            Cs[i * DIM + o1] = acc1[v];
        }
    }
    __syncthreads();
    // LayerNorm over 512 outputs: 8 threads per row
    int row = t >> 3, seg = t & 7;
    float s = 0.f;
    for (int j = seg * 64; j < seg * 64 + 64; ++j) s += Cs[row * DIM + j];
    red[row * 8 + seg] = s; __syncthreads();
    float mean = 0.f;
    for (int q = 0; q < 8; ++q) mean += red[row * 8 + q];
    mean *= (1.f / DIM);
    __syncthreads();
    float s2 = 0.f;
    for (int j = seg * 64; j < seg * 64 + 64; ++j) { float d = Cs[row * DIM + j] - mean; s2 += d * d; }
    red[row * 8 + seg] = s2; __syncthreads();
    float var = 0.f;
    for (int q = 0; q < 8; ++q) var += red[row * 8 + q];
    float rstd = rsqrtf(var * (1.f / DIM) + 1e-5f);
    float* dst = out + (size_t)(m0 + row) * DIM;
    for (int j = seg * 64; j < seg * 64 + 64; ++j)
        dst[j] = (Cs[row * DIM + j] - mean) * rstd * gout[j];
}

// ---------------- host launcher ---------------------------------------------
extern "C" void kernel_launch(void* const* d_in, const int* in_sizes, int n_in,
                              void* d_out, int out_size, void* d_ws, size_t ws_size,
                              hipStream_t stream) {
    const float* x     = (const float*)d_in[0];
    const float* ctx   = (const float*)d_in[1];
    const float* ng    = (const float*)d_in[2];
    const float* wq    = (const float*)d_in[3];
    const float* wkv   = (const float*)d_in[4];
    const float* nkv   = (const float*)d_in[5];
    const float* wout  = (const float*)d_in[6];
    const float* gout  = (const float*)d_in[7];
    float* out = (float*)d_out;

    char* wsp = (char*)d_ws; size_t off = 0;
    auto carve = [&](size_t bytes) { char* p = wsp + off; off += (bytes + 255) & ~(size_t)255; return p; };
    bf16_t* xn    = (bf16_t*)carve((size_t)MROWS * DIM * 2);
    bf16_t* cn    = (bf16_t*)carve((size_t)MROWS * DIM * 2);
    bf16_t* wqb   = (bf16_t*)carve((size_t)DIM * DIM * 2);
    bf16_t* wkvb  = (bf16_t*)carve((size_t)2 * DIM * DIM * 2);
    bf16_t* woutb = (bf16_t*)carve((size_t)DIM * DIM * 2);
    bf16_t* q_t   = (bf16_t*)carve((size_t)32 * NN * RD * 2);
    bf16_t* k_t   = (bf16_t*)carve((size_t)32 * JT * RD * 2);
    bf16_t* v_t   = (bf16_t*)carve((size_t)32 * RD * JT * 2);
    bf16_t* omid  = (bf16_t*)carve((size_t)MROWS * DIM * 2);

    k_cast_w<<<2048, 256, 0, stream>>>(wq, wkv, wout, wqb, wkvb, woutb);
    k_init_kv<<<(32 * RD * 64 + 255) / 256, 256, 0, stream>>>(nkv, k_t, v_t);
    k_ln_cast<<<dim3(MROWS, 2), 128, 0, stream>>>(x, ctx, ng, xn, cn);
    k_qproj<<<dim3(MROWS / 16, 4), 128, 0, stream>>>(xn, wqb, q_t);
    k_kvproj<<<dim3(MROWS / 16, 8), 128, 0, stream>>>(cn, wkvb, k_t, v_t);

    size_t attn_lds = (size_t)32 * RD * 2 + (size_t)32 * JT * 4 + (size_t)32 * JT * 2; // 251904 B
    (void)hipFuncSetAttribute((const void*)k_attn,
                              hipFuncAttributeMaxDynamicSharedMemorySize, (int)attn_lds);
    k_attn<<<dim3(NN / 32, 32), 128, attn_lds, stream>>>(q_t, k_t, v_t, omid);

    k_outproj<<<MROWS / 16, 128, 0, stream>>>(omid, woutb, gout, out);
    (void)in_sizes; (void)n_in; (void)out_size; (void)ws_size;
}
